// FilterbankFeatures_14525579395585
// MI455X (gfx1250) — matrix operand order repaired and checked by reading, not compiled, not run on hardware
//
#include <hip/hip_runtime.h>
#include <hip/hip_bf16.h>
#include <math.h>

typedef __attribute__((ext_vector_type(16))) _Float16 v16h;
typedef __attribute__((ext_vector_type(8)))  float    v8f;

#define T_LEN     133600
#define HOP       80
#define N_FRAMES  1671
#define MAX_LEN   1680
#define N_FTILES  105          // 105*16 = 1680 >= 1671
#define FREQ      129
#define N_NTILES  9            // freq padded 129 -> 144
#define N_KTILES  5            // windowed K: 160 nonzero samples / 32
#define N_MELS    64
#define N_MTILES  4            // 64 / 16
#define N_K2TILES 5            // freq-K padded 129 -> 160
#define B_SIZE    128
#define SPAN      1457         // raw samples staged per tile (incl. s-1)
#define PW_STRIDE 17           // 160x16 power tile, padded row stride

// d_ws layout in _Float16 units (fragment = 32 lanes * 16 halves = 512)
#define WC_OFF 0                        // 9*5 = 45 frags (window folded in)
#define WS_OFF (45 * 512)               // 45 frags
#define FB_OFF (2 * 45 * 512)           // 4*5 = 20 frags

// ---------------------------------------------------------------------------
// Kernel 1: bake windowed-DFT (w*cos / w*sin) and mel-filterbank matrices into
// WMMA fragment order.
//  B-fragment (32x16 f16): lane(h=lane>>4, r=lane&15), half j holds
//    B[k0 + 16h + j][n0 + r].
//  A-fragment (16x32 f16): lane(h,r), half j holds
//    A[m0 + r][k0 + kl], kl = (j<8) ? 8h+j : 8h+j+8.
// ---------------------------------------------------------------------------
__global__ void init_frags(const float* __restrict__ window,
                           const float* __restrict__ fb,
                           _Float16* __restrict__ ws) {
    int idx = blockIdx.x * 256 + threadIdx.x;
    if (idx < 45 * 512) {
        int frag = idx >> 9, e = idx & 511;
        int lane = e >> 4, j = e & 15;
        int h = lane >> 4, r = lane & 15;
        int nt = frag / N_KTILES, kt = frag % N_KTILES;
        int kp = kt * 32 + 16 * h + j;        // windowed-K index (0..159)
        int ns = 48 + kp;                     // actual FFT sample (48..207)
        int n  = nt * 16 + r;                 // freq bin (0..143)
        int kn = (ns * n) & 255;              // exact modular reduction
        float th = (float)kn * 0.0245436926f; // 2*pi/256
        float w  = window[kp];                // fold Hamming window into B
        ws[WC_OFF + idx] = (_Float16)(w * cosf(th));
        ws[WS_OFF + idx] = (_Float16)(w * sinf(th));
    } else if (idx < 45 * 512 + 20 * 512) {
        int t = idx - 45 * 512;
        int frag = t >> 9, e = t & 511;
        int lane = e >> 4, j = e & 15;
        int h = lane >> 4, r = lane & 15;
        int mt = frag / N_K2TILES, k2t = frag % N_K2TILES;
        int m = mt * 16 + r;                               // mel row (0..63)
        int kl = (j < 8) ? (8 * h + j) : (8 * h + j + 8);
        int k = k2t * 32 + kl;                             // freq-K (0..159)
        float v = (k < FREQ) ? fb[m * FREQ + k] : 0.0f;
        ws[FB_OFF + t] = (_Float16)v;
    }
}

// ---------------------------------------------------------------------------
// Kernel 2: one wave per (batch, 16-frame tile).
// Stage A: async-stage raw x span -> LDS (ASYNCcnt), preemph+reflect -> LDS.
// Stage B: windowed DFT via WMMA f16 (f32 accum): power -> LDS (160x16).
// Stage C: mel via WMMA, log -> d_out (staging buffer, fully rewritten).
// ---------------------------------------------------------------------------
__global__ void __launch_bounds__(32)
fbank_gemm(const float* __restrict__ x, const _Float16* __restrict__ ws,
           float* __restrict__ out) {
    __shared__ float raw_s[SPAN];                  // raw x window [lo, lo+1456]
    __shared__ float xp_s[1456];                   // preemph + reflect-padded
    __shared__ float pw_s[160 * PW_STRIDE];        // power [freqK][frame]

    const int lane = threadIdx.x;
    const int h = lane >> 4, r = lane & 15;
    const int b = blockIdx.y;
    const int ftile = blockIdx.x;

    const float* xb = x + (size_t)b * T_LEN;
    const int base = ftile * 1280 - 128;
    // one contiguous raw span covers every (reflected) index this tile touches
    int lo = base - 1;
    lo = (lo < 0) ? 0 : lo;
    lo = (lo > T_LEN - SPAN) ? (T_LEN - SPAN) : lo;

    // ---- async global -> LDS staging of the raw span (ASYNCcnt path) ----
    {
        unsigned lds_base = (unsigned)(size_t)&raw_s[0];
        for (int i = lane; i < SPAN; i += 32) {
            unsigned long long ga = (unsigned long long)(size_t)(xb + lo + i);
            unsigned la = lds_base + 4u * i;
            asm volatile("global_load_async_to_lds_b32 %0, %1, off"
                         :: "v"(la), "v"(ga) : "memory");
        }
        asm volatile("s_wait_asynccnt 0" ::: "memory");
    }
    // zero pad-rows 144..159 of the power tile
    for (int i = lane; i < 16 * PW_STRIDE; i += 32) pw_s[144 * PW_STRIDE + i] = 0.0f;
    __syncthreads();

    // ---- pre-emphasis + reflect padding from the LDS raw span ----
    for (int s = lane; s < 1456; s += 32) {
        int j = base + s;
        int si = (j < 0) ? -j : ((j >= T_LEN) ? (2 * T_LEN - 2 - j) : j);
        float p = (si == 0) ? raw_s[0]
                            : (raw_s[si - lo] - 0.97f * raw_s[si - lo - 1]);
        xp_s[s] = p;
    }
    __syncthreads();

    // ---- pack A fragments: raw frames over the nonzero window (K=160) ----
    v16h af[N_KTILES];
#pragma unroll
    for (int kt = 0; kt < N_KTILES; ++kt) {
#pragma unroll
        for (int j = 0; j < 16; ++j) {
            int kl = 16 * h + j;
            af[kt][j] = (_Float16)xp_s[r * HOP + 48 + kt * 32 + kl];
        }
    }

    const v16h* wc  = (const v16h*)(ws + WC_OFF);
    const v16h* wsn = (const v16h*)(ws + WS_OFF);
    const v16h* fbf = (const v16h*)(ws + FB_OFF);

    // ---- DFT: D[frame][freq] = frames x {w*cos, w*sin}; power -> LDS ----
    for (int nt = 0; nt < N_NTILES; ++nt) {
        v8f are = {};
        v8f aim = {};
#pragma unroll
        for (int kt = 0; kt < N_KTILES; ++kt) {
            v16h bc = wc[(nt * N_KTILES + kt) * 32 + lane];
            v16h bs = wsn[(nt * N_KTILES + kt) * 32 + lane];
            are = __builtin_amdgcn_wmma_f32_16x16x32_f16(
                false, af[kt], false, bc, (short)0, are, false, false);
            aim = __builtin_amdgcn_wmma_f32_16x16x32_f16(
                false, af[kt], false, bs, (short)0, aim, false, false);
        }
        int fq = nt * 16 + r;           // freq bin = D's N = lane.r
#pragma unroll
        for (int i = 0; i < 8; ++i) {   // frame = D's M = i + 8h
            float p = are[i] * are[i] + aim[i] * aim[i] + 1e-10f;
            pw_s[fq * PW_STRIDE + i + 8 * h] = p;
        }
    }
    __syncthreads();

    // ---- mel: D[mel][frame] = fb x power ----
    v8f acc[N_MTILES];
#pragma unroll
    for (int mt = 0; mt < N_MTILES; ++mt) acc[mt] = (v8f){};
    for (int k2t = 0; k2t < N_K2TILES; ++k2t) {
        v16h b2;
#pragma unroll
        for (int j = 0; j < 16; ++j)
            b2[j] = (_Float16)pw_s[(k2t * 32 + ((j < 8) ? (8 * h + j) : (8 * h + j + 8))) * PW_STRIDE + r];
#pragma unroll
        for (int mt = 0; mt < N_MTILES; ++mt) {
            v16h a2 = fbf[(mt * N_K2TILES + k2t) * 32 + lane];
            acc[mt] = __builtin_amdgcn_wmma_f32_16x16x32_f16(
                false, a2, false, b2, (short)0, acc[mt], false, false);
        }
    }

    const int frame = ftile * 16 + r;   // D's N
    if (frame < N_FRAMES) {
#pragma unroll
        for (int mt = 0; mt < N_MTILES; ++mt)
#pragma unroll
            for (int i = 0; i < 8; ++i) {
                int mel = mt * 16 + i + 8 * h;   // D's M
                out[((size_t)(b * N_MELS + mel)) * MAX_LEN + frame] =
                    logf(acc[mt][i] + 1e-20f);
            }
    }
}

// ---------------------------------------------------------------------------
// Kernel 3: per-(b,mel)-row masked mean/var normalization, in place.
// ---------------------------------------------------------------------------
__global__ void __launch_bounds__(256)
norm_rows(const int* __restrict__ x_lens, float* __restrict__ out) {
    __shared__ float red[256];
    const int row = blockIdx.x;            // b*64 + mel
    const int b = row >> 6;
    float* p = out + (size_t)row * MAX_LEN;
    const int seq = x_lens[b] / HOP + 1;   // == floor(len/80 + 1)
    const int tid = threadIdx.x;

    float vals[7];
    float s = 0.0f;
#pragma unroll
    for (int k = 0; k < 7; ++k) {
        int t = tid + k * 256;
        float v = (t < seq) ? p[t] : 0.0f;
        vals[k] = v;
        s += v;
    }
    red[tid] = s;
    __syncthreads();
    for (int o = 128; o > 0; o >>= 1) {
        if (tid < o) red[tid] += red[tid + o];
        __syncthreads();
    }
    const float mean = red[0] / (float)seq;
    __syncthreads();

    float ss = 0.0f;
#pragma unroll
    for (int k = 0; k < 7; ++k) {
        int t = tid + k * 256;
        if (t < seq) { float d = vals[k] - mean; ss += d * d; }
    }
    red[tid] = ss;
    __syncthreads();
    for (int o = 128; o > 0; o >>= 1) {
        if (tid < o) red[tid] += red[tid + o];
        __syncthreads();
    }
    const float var = red[0] / ((float)seq - 1.0f);
    const float inv = 1.0f / (sqrtf(var) + 1e-12f);

#pragma unroll
    for (int k = 0; k < 7; ++k) {
        int t = tid + k * 256;
        if (t < MAX_LEN)
            p[t] = (t < seq) ? (vals[k] - mean) * inv : 0.0f;
    }
}

__global__ void seq_kernel(const int* __restrict__ x_lens, int* __restrict__ tail) {
    int b = threadIdx.x;
    if (b < B_SIZE) tail[b] = x_lens[b] / HOP + 1;
}

// ---------------------------------------------------------------------------
extern "C" void kernel_launch(void* const* d_in, const int* in_sizes, int n_in,
                              void* d_out, int out_size, void* d_ws, size_t ws_size,
                              hipStream_t stream) {
    const float* x      = (const float*)d_in[0];   // (128, 133600)
    const float* window = (const float*)d_in[1];   // (160,)
    const float* fb     = (const float*)d_in[2];   // (64, 129)
    const int*   x_lens = (const int*)d_in[3];     // (128,)
    float* out = (float*)d_out;                    // 128*64*1680 f32 + 128 i32
    _Float16* ws = (_Float16*)d_ws;                // ~113 KB of fragment tables

    // 1) bake windowed-DFT + mel matrices in WMMA-fragment order (45+45+20)
    init_frags<<<((45 * 512 + 20 * 512) + 255) / 256, 256, 0, stream>>>(
        window, fb, ws);

    // 2) async-staged framing + DFT + mel-log via WMMA: 1 wave per (b, tile)
    fbank_gemm<<<dim3(N_FTILES, B_SIZE), 32, 0, stream>>>(x, ws, out);

    // 3) masked mean/var normalization per (b, mel) row, in place
    norm_rows<<<B_SIZE * N_MELS, 256, 0, stream>>>(x_lens, out);

    // 4) seq_len tail (int32 bits appended after the feature tensor)
    seq_kernel<<<1, 128, 0, stream>>>(
        x_lens, (int*)(out + (size_t)B_SIZE * N_MELS * MAX_LEN));
}
